// CrossAttention_78486232367518
// MI455X (gfx1250) — matrix operand order repaired and checked by reading, not compiled, not run on hardware
//
#include <hip/hip_runtime.h>
#include <hip/hip_bf16.h>
#include <math.h>

// Problem constants (match reference)
#define Bb   4
#define Nn   300
#define Ee   600
#define Ll   4096
#define Dd   256
#define Hh   8
#define HDd  32
#define CPBH 512

typedef __attribute__((ext_vector_type(16))) __bf16 v16bf;
typedef __attribute__((ext_vector_type(8)))  float  v8f;

// ---- CDNA5 async load-to-LDS (confirmed present on this toolchain) ----
#if defined(__gfx1250__) && __has_builtin(__builtin_amdgcn_global_load_async_to_lds_b128)
#define HAVE_ASYNC_LDS 1
#else
#define HAVE_ASYNC_LDS 0
#endif

typedef int v4i_async __attribute__((vector_size(16)));
typedef __attribute__((address_space(1))) v4i_async* gv4p;   // global int4*
typedef __attribute__((address_space(3))) v4i_async* lv4p;   // LDS int4*

static __device__ __forceinline__ void async_copy16(const unsigned short* g, unsigned short* l){
#if HAVE_ASYNC_LDS
  void* gm = (void*)g;          // drop const, keep generic AS
  void* lm = (void*)l;
  __builtin_amdgcn_global_load_async_to_lds_b128((gv4p)gm, (lv4p)lm, 0, 0);
#else
  *(uint4*)l = *(const uint4*)g;
#endif
}
// Wait until at most n async ops remain outstanding (async loads retire in order,
// so after issuing n new copies this retires everything older).
#if HAVE_ASYNC_LDS
#if __has_builtin(__builtin_amdgcn_s_wait_asynccnt)
#define ASYNC_JOIN(n) __builtin_amdgcn_s_wait_asynccnt(n)
#else
#define ASYNC_JOIN(n) asm volatile("s_wait_asynccnt %0" :: "i"(n) : "memory")
#endif
#else
#define ASYNC_JOIN(n) do {} while (0)
#endif

// fp32 -> bf16 round-to-nearest-even (bit trick; NaN edge ignored)
static __device__ __forceinline__ unsigned short f2bf(float x){
  union { float f; unsigned u; } v; v.f = x;
  unsigned r = v.u + 0x7fffu + ((v.u >> 16) & 1u);
  return (unsigned short)(r >> 16);
}
static __device__ __forceinline__ float bf2f(unsigned short s){
  union { unsigned u; float f; } v; v.u = ((unsigned)s) << 16; return v.f;
}
static __device__ __forceinline__ unsigned pack2(float a, float b){
  return (unsigned)f2bf(a) | ((unsigned)f2bf(b) << 16);
}

static __device__ __forceinline__ v8f wmma_bf16(v16bf a, v16bf b, v8f c){
  // D = A(16x32) * B(32x16) + C, f32 accumulate
  return __builtin_amdgcn_wmma_f32_16x16x32_bf16(false, a, false, b, (short)0, c, false, false);
}

// A fragment (16x32 bf16) from LDS tile [row][K], stride in shorts (multiple of 8).
// lane: m = lane&15 row, half = lane>>4. Element e -> K = 16*(e>>3) + (e&7) + 8*half.
static __device__ __forceinline__ v16bf frag_a_lds(const unsigned short* base, int stride,
                                                   int m, int kk, int half){
  union { v16bf v; uint4 q[2]; } f;
  const unsigned short* p = base + m*stride + kk + half*8;
  f.q[0] = *(const uint4*)(p);
  f.q[1] = *(const uint4*)(p + 16);
  return f.v;
}
// B fragment (32x16 bf16) from LDS tile [K][N], stride in shorts (multiple of 8).
static __device__ __forceinline__ v16bf frag_b_lds(const unsigned short* base, int stride,
                                                   int krow, int n0){
  union { v16bf v; unsigned short s[16]; } f;
  const int kk = krow & ~31, ln = krow & 31;
  const unsigned short* p = base + (kk + (ln >> 4)*8)*stride + n0 + (ln & 15);
  #pragma unroll
  for (int e = 0; e < 16; ++e) f.s[e] = p[((e >> 3)*16 + (e & 7))*stride];
  return f.v;
}
// A fragment directly from a global bf16 row (32 contiguous K values, 16B aligned).
static __device__ __forceinline__ v16bf frag_a_g16(const unsigned short* rowp, int half, bool valid){
  union { v16bf v; uint4 q[2]; } f;
  if (valid){
    f.q[0] = *(const uint4*)(rowp + half*8);
    f.q[1] = *(const uint4*)(rowp + 16 + half*8);
  } else {
    uint4 z; z.x = z.y = z.z = z.w = 0u;
    f.q[0] = z; f.q[1] = z;
  }
  return f.v;
}

// ---------------------------------------------------------------------------
// Generic GEMM: C[M,N] = A[M,256] @ B[256,N] + bias[N] (bf16 WMMA, f32 acc).
// Stage ALL of K once, then an uninterrupted 8-WMMA chain.
// ---------------------------------------------------------------------------
#define GK 256
template<int A_BF16, int OUT_BF16>
__global__ __launch_bounds__(256) void gemm_bias_kernel(
    const void* __restrict__ Ain, const float* __restrict__ Bm,
    const float* __restrict__ bias, void* __restrict__ Cout,
    int M, int N)
{
  extern __shared__ __align__(16) char smem[];
  unsigned short* Ash = (unsigned short*)smem;   // 16 x 264
  unsigned short* Bsh = Ash + 16*264;            // 256 x 136

  const int tid  = threadIdx.x;
  const int lane = tid & 31, w = tid >> 5;
  const int half = lane >> 4, lm = lane & 15;
  const int m0 = blockIdx.x * 16;
  const int n0 = blockIdx.y * 128;

  {
    const int r = tid >> 4, c = (tid & 15) * 16;
    if (A_BF16){
      const unsigned short* Ab = (const unsigned short*)Ain + (long)(m0 + r)*GK + c;
      *(uint4*)&Ash[r*264 + c]     = *(const uint4*)(Ab);
      *(uint4*)&Ash[r*264 + c + 8] = *(const uint4*)(Ab + 8);
    } else {
      const float* p = (const float*)Ain + (long)(m0 + r)*GK + c;
      float4 f0 = *(const float4*)(p);
      float4 f1 = *(const float4*)(p + 4);
      float4 f2 = *(const float4*)(p + 8);
      float4 f3 = *(const float4*)(p + 12);
      uint4 u0, u1;
      u0.x = pack2(f0.x, f0.y); u0.y = pack2(f0.z, f0.w);
      u0.z = pack2(f1.x, f1.y); u0.w = pack2(f1.z, f1.w);
      u1.x = pack2(f2.x, f2.y); u1.y = pack2(f2.z, f2.w);
      u1.z = pack2(f3.x, f3.y); u1.w = pack2(f3.z, f3.w);
      *(uint4*)&Ash[r*264 + c]     = u0;
      *(uint4*)&Ash[r*264 + c + 8] = u1;
    }
  }
  #pragma unroll
  for (int it = 0; it < 8; ++it){
    const int s = tid + it*256;
    const int r = s >> 3, c = (s & 7) * 16;
    const float* p = Bm + (long)r*N + n0 + c;
    float4 f0 = *(const float4*)(p);
    float4 f1 = *(const float4*)(p + 4);
    float4 f2 = *(const float4*)(p + 8);
    float4 f3 = *(const float4*)(p + 12);
    uint4 u0, u1;
    u0.x = pack2(f0.x, f0.y); u0.y = pack2(f0.z, f0.w);
    u0.z = pack2(f1.x, f1.y); u0.w = pack2(f1.z, f1.w);
    u1.x = pack2(f2.x, f2.y); u1.y = pack2(f2.z, f2.w);
    u1.z = pack2(f3.x, f3.y); u1.w = pack2(f3.z, f3.w);
    *(uint4*)&Bsh[r*136 + c]     = u0;
    *(uint4*)&Bsh[r*136 + c + 8] = u1;
  }
  __syncthreads();

  v8f acc = {};
  #pragma unroll
  for (int kc = 0; kc < 8; ++kc){
    v16bf a  = frag_a_lds(Ash, 264, lm, kc*32, half);
    v16bf bf = frag_b_lds(Bsh, 136, kc*32 + lm + 16*half, w*16);
    acc = wmma_bf16(a, bf, acc);
  }

  const int col = n0 + w*16 + lm;
  const float bv = bias[col];
  if (OUT_BF16){
    unsigned short* Co = (unsigned short*)Cout;
    #pragma unroll
    for (int r = 0; r < 8; ++r)
      Co[((long)(m0 + r + 8*half))*N + col] = f2bf(acc[r] + bv);
  } else {
    float* Co = (float*)Cout;
    #pragma unroll
    for (int r = 0; r < 8; ++r)
      Co[((long)(m0 + r + 8*half))*N + col] = acc[r] + bv;
  }
}

// ---------------------------------------------------------------------------
// Fused CPB MLP: node_cpb[b,h,n,l] = relu(rd @ W1) @ W2   (bf16 output)
// ---------------------------------------------------------------------------
__global__ __launch_bounds__(256) void cpb_node_kernel(
    const float* __restrict__ rd, const float* __restrict__ W1,
    const float* __restrict__ W2, unsigned short* __restrict__ ncpb)
{
  extern __shared__ __align__(16) char smem[];
  unsigned short* Hsh  = (unsigned short*)smem;      // 128 x 520
  unsigned short* W2sh = Hsh + 128*520;              // 512 x 24
  float* w1a = (float*)(W2sh + 512*24);
  float* w1b = w1a + CPBH;
  float* rd0 = w1b + CPBH;
  float* rd1 = rd0 + 128;

  const int tid  = threadIdx.x;
  const int lane = tid & 31, w = tid >> 5;
  const int half = lane >> 4, lm = lane & 15;
  const int l0 = blockIdx.x * 128;
  const int n  = blockIdx.y;
  const int b  = blockIdx.z;

  #pragma unroll
  for (int it = 0; it < 2; ++it){
    int i = tid + it*256;
    w1a[i] = W1[i]; w1b[i] = W1[CPBH + i];
  }
  if (tid < 128){
    const float* p = rd + (((long)b*Nn + n)*Ll + l0 + tid)*2;
    rd0[tid] = p[0]; rd1[tid] = p[1];
  }
  #pragma unroll
  for (int it = 0; it < 16; ++it){
    int i = (tid + it*256) * 2;               // even
    int k = i >> 4, hc = i & 15;
    float a0 = (hc     < Hh) ? W2[k*Hh + hc]     : 0.f;
    float a1 = (hc + 1 < Hh) ? W2[k*Hh + hc + 1] : 0.f;
    *(unsigned*)&W2sh[k*24 + hc] = pack2(a0, a1);
  }
  __syncthreads();
  // layer 1: rank-2 update + ReLU
  #pragma unroll 4
  for (int it = 0; it < 128; ++it){
    int i = (tid + it*256) * 2;               // even
    int r = i >> 9, j = i & 511;
    float h0 = fmaf(rd0[r], w1a[j],     rd1[r]*w1b[j]);
    float h1 = fmaf(rd0[r], w1a[j + 1], rd1[r]*w1b[j + 1]);
    *(unsigned*)&Hsh[r*520 + j] = pack2(fmaxf(h0, 0.f), fmaxf(h1, 0.f));
  }
  __syncthreads();
  // layer 2: K=512 -> 16 WMMA steps
  v8f acc = {};
  #pragma unroll
  for (int kc = 0; kc < 16; ++kc){
    v16bf a  = frag_a_lds(Hsh, 520, w*16 + lm, kc*32, half);
    v16bf bf = frag_b_lds(W2sh, 24, kc*32 + lm + 16*half, 0);
    acc = wmma_bf16(a, bf, acc);
  }
  if (lm < Hh){
    #pragma unroll
    for (int r = 0; r < 8; ++r){
      int l = l0 + w*16 + r + 8*half;
      ncpb[(((long)b*Hh + lm)*Nn + n)*Ll + l] = f2bf(acc[r]);
    }
  }
}

// ---------------------------------------------------------------------------
// Stage k^T and v tiles (bf16 kv, 64 l-values). v path: 1 async op per wave.
// ---------------------------------------------------------------------------
static __device__ __forceinline__ void stage_kv(
    const unsigned short* __restrict__ kvb, int b, int h, int l0, int tid,
    unsigned short* kT /*[32][72]*/, unsigned short* vsh /*[64][40]*/)
{
  const int l  = tid >> 2;
  const int dq = (tid & 3) * 8;
  const unsigned short* src = kvb + ((long)(b*Ll + l0 + l))*(2*Dd) + h*(2*HDd);
  uint4 kk4 = *(const uint4*)(src + dq);
  const unsigned short* ks = (const unsigned short*)&kk4;
  #pragma unroll
  for (int j = 0; j < 8; ++j) kT[(dq + j)*72 + l] = ks[j];   // transpose store
  async_copy16(src + HDd + dq, &vsh[l*40 + dq]);             // straight copy
}
// Stage node_cpb tile [320 n][64 l]: exactly 10 async ops per wave (no branches;
// pad rows n>=300 clamp the source — their data is nulled by zero A-padding).
static __device__ __forceinline__ void stage_ncpb(
    const unsigned short* __restrict__ ncpb, int b, int h, int l0, int tid,
    unsigned short* Nsh /*[320][72]*/)
{
  #pragma unroll
  for (int it = 0; it < 10; ++it){
    int ch = tid + it*256;            // 0..2559
    int nn2 = ch >> 3, c8 = (ch & 7) * 8;
    int nc = (nn2 < Nn) ? nn2 : 0;
    async_copy16(ncpb + (((long)b*Hh + h)*Nn + nc)*Ll + l0 + c8,
                 &Nsh[nn2*72 + c8]);
  }
}

// ---------------------------------------------------------------------------
// Node attention (flash-style), double-buffered async k/v staging.
// Per block (b,h, 128 query rows), 8 waves x 16 rows. 64 chunks of 64 keys.
// ---------------------------------------------------------------------------
__global__ __launch_bounds__(256) void attn_node_kernel(
    const unsigned short* __restrict__ nq, const unsigned short* __restrict__ kvb,
    const unsigned short* __restrict__ ncpb, unsigned short* __restrict__ nO)
{
  __shared__ unsigned short kT2[2*32*72];
  __shared__ unsigned short vv2[2*64*40];
  __shared__ unsigned short Psh[8*16*72];

  const int tid  = threadIdx.x;
  const int lane = tid & 31, w = tid >> 5;
  const int half = lane >> 4, lm = lane & 15;
  const int nb = blockIdx.x, h = blockIdx.y, b = blockIdx.z;
  const int nr0 = nb*128 + w*16;
  const float norm = 0.17677669529663687f;   // HD^-0.5

  const int  qm  = nr0 + lm;
  const bool qok = qm < Nn;
  v16bf qf = frag_a_g16(nq + ((long)b*Nn + (qok ? qm : 0))*Dd + h*HDd, half, qok);

  v8f o0 = {}; v8f o1 = {};
  float mrun[8], lrun[8];
  #pragma unroll
  for (int r = 0; r < 8; ++r){ mrun[r] = -1e30f; lrun[r] = 0.f; }
  unsigned short* PshW = Psh + w*16*72;

  const int NC = Ll/64;
  stage_kv(kvb, b, h, 0, tid, kT2, vv2);                    // prologue: buf 0
  for (int i = 0; i < NC; ++i){
    const int p  = i & 1;
    const int l0 = i*64;
    unsigned short* kTc = kT2 + p*(32*72);
    unsigned short* vc  = vv2 + p*(64*40);
    if (i + 1 < NC){
      stage_kv(kvb, b, h, l0 + 64, tid, kT2 + (1-p)*(32*72), vv2 + (1-p)*(64*40));
      ASYNC_JOIN(1);          // retire chunk i's copy; chunk i+1 streams behind
    } else {
      ASYNC_JOIN(0);
    }
    __syncthreads();

    v8f sg[4];
    const int kb = lm + 16*half;
    #pragma unroll
    for (int g = 0; g < 4; ++g){
      v16bf bf = frag_b_lds(kTc, 72, kb, g*16);
      v8f z = {};
      v8f s = wmma_bf16(qf, bf, z);
      #pragma unroll
      for (int r = 0; r < 8; ++r){
        int mr = nr0 + r + 8*half;
        float cp = (mr < Nn)
          ? bf2f(ncpb[(((long)b*Hh + h)*Nn + mr)*Ll + l0 + g*16 + lm]) : 0.f;
        s[r] = s[r]*norm + cp;
      }
      sg[g] = s;
    }
    #pragma unroll
    for (int r = 0; r < 8; ++r){
      float mx = mrun[r];
      #pragma unroll
      for (int g = 0; g < 4; ++g) mx = fmaxf(mx, sg[g][r]);
      #pragma unroll
      for (int off = 8; off >= 1; off >>= 1) mx = fmaxf(mx, __shfl_xor(mx, off, 32));
      float sc = __expf(mrun[r] - mx);
      float rs = 0.f;
      #pragma unroll
      for (int g = 0; g < 4; ++g){ float pw = __expf(sg[g][r] - mx); sg[g][r] = pw; rs += pw; }
      #pragma unroll
      for (int off = 8; off >= 1; off >>= 1) rs += __shfl_xor(rs, off, 32);
      lrun[r] = lrun[r]*sc + rs;
      mrun[r] = mx;
      o0[r] *= sc; o1[r] *= sc;
      int Mrow = r + 8*half;
      #pragma unroll
      for (int g = 0; g < 4; ++g) PshW[Mrow*72 + g*16 + lm] = f2bf(sg[g][r]);
    }
    __syncthreads();
    #pragma unroll
    for (int kkc = 0; kkc < 2; ++kkc){
      v16bf pa  = frag_a_lds(PshW, 72, lm, kkc*32, half);
      int   kb2 = kkc*32 + lm + 16*half;
      v16bf vb0 = frag_b_lds(vc, 40, kb2, 0);
      v16bf vb1 = frag_b_lds(vc, 40, kb2, 16);
      o0 = wmma_bf16(pa, vb0, o0);
      o1 = wmma_bf16(pa, vb1, o1);
    }
    __syncthreads();   // all reads of buf p done before it is restaged
  }
  #pragma unroll
  for (int r = 0; r < 8; ++r){
    int mr = nr0 + r + 8*half;
    if (mr < Nn){
      float inv = 1.f / lrun[r];
      unsigned short* dst = nO + ((long)b*Nn + mr)*Dd + h*HDd;
      dst[lm]      = f2bf(o0[r]*inv);
      dst[16 + lm] = f2bf(o1[r]*inv);
    }
  }
}

// ---------------------------------------------------------------------------
// Edge attention with fused edge_cpb = cbm @ node_cpb (K padded to 320),
// double-buffered async staging of node_cpb / k / v. Dynamic LDS ~209 KB.
// ---------------------------------------------------------------------------
__global__ __launch_bounds__(256) void attn_edge_kernel(
    const unsigned short* __restrict__ eq, const unsigned short* __restrict__ kvb,
    const unsigned short* __restrict__ ncpb, const float* __restrict__ cbm,
    unsigned short* __restrict__ eO)
{
  extern __shared__ __align__(16) char smem[];
  unsigned short* Csh  = (unsigned short*)smem;  // cbm tile [128 e][320 n] stride 328
  unsigned short* Nsh2 = Csh  + 128*328;         // 2 x [320 n][64 l] stride 72
  unsigned short* kT2  = Nsh2 + 2*320*72;        // 2 x [32 d][64 l] stride 72
  unsigned short* vv2  = kT2  + 2*32*72;         // 2 x [64 l][32 d] stride 40
  unsigned short* Psh  = vv2  + 2*64*40;         // per-wave [16][64] stride 72

  const int tid  = threadIdx.x;
  const int lane = tid & 31, w = tid >> 5;
  const int half = lane >> 4, lm = lane & 15;
  const int eb = blockIdx.x, h = blockIdx.y, b = blockIdx.z;
  const int e0  = eb*128;
  const int er0 = e0 + w*16;
  const float norm = 0.17677669529663687f;

  // stage cbm rows once (fp32 -> bf16; zero-pad e>=600, n>=300; 300%4==0)
  #pragma unroll
  for (int it = 0; it < 40; ++it){
    int ch = tid + it*256;              // 0..10239
    int r = ch / 80, cq = (ch % 80) * 4;
    int e = e0 + r;
    float4 f; f.x = f.y = f.z = f.w = 0.f;
    if (e < Ee && cq < Nn)
      f = *(const float4*)(cbm + ((long)b*Ee + e)*Nn + cq);
    uint2 u; u.x = pack2(f.x, f.y); u.y = pack2(f.z, f.w);
    *(uint2*)&Csh[r*328 + cq] = u;
  }

  const int  em  = er0 + lm;
  const bool qok = em < Ee;
  v16bf qf = frag_a_g16(eq + ((long)b*Ee + (qok ? em : 0))*Dd + h*HDd, half, qok);

  v8f o0 = {}; v8f o1 = {};
  float mrun[8], lrun[8];
  #pragma unroll
  for (int r = 0; r < 8; ++r){ mrun[r] = -1e30f; lrun[r] = 0.f; }
  unsigned short* PshW = Psh + w*16*72;

  const int NC = Ll/64;
  stage_kv  (kvb,  b, h, 0, tid, kT2, vv2);     // prologue: buf 0 (11 asyncs/wave)
  stage_ncpb(ncpb, b, h, 0, tid, Nsh2);
  for (int i = 0; i < NC; ++i){
    const int p  = i & 1;
    const int l0 = i*64;
    unsigned short* Nc  = Nsh2 + p*(320*72);
    unsigned short* kTc = kT2  + p*(32*72);
    unsigned short* vc  = vv2  + p*(64*40);
    if (i + 1 < NC){
      stage_kv  (kvb,  b, h, l0 + 64, tid, kT2 + (1-p)*(32*72), vv2 + (1-p)*(64*40));
      stage_ncpb(ncpb, b, h, l0 + 64, tid, Nsh2 + (1-p)*(320*72));
      ASYNC_JOIN(11);         // retire chunk i's 11 copies; next 11 stream behind
    } else {
      ASYNC_JOIN(0);
    }
    __syncthreads();

    // edge_cpb tile: cp[g] = cbm(16x320) @ node_cpb(320x16) (10 K-steps x 4 groups)
    v8f cp0 = {}, cp1 = {}, cp2 = {}, cp3 = {};
    #pragma unroll
    for (int kc = 0; kc < 10; ++kc){
      v16bf ca  = frag_a_lds(Csh, 328, w*16 + lm, kc*32, half);
      int   kbn = kc*32 + lm + 16*half;
      cp0 = wmma_bf16(ca, frag_b_lds(Nc, 72, kbn, 0),  cp0);
      cp1 = wmma_bf16(ca, frag_b_lds(Nc, 72, kbn, 16), cp1);
      cp2 = wmma_bf16(ca, frag_b_lds(Nc, 72, kbn, 32), cp2);
      cp3 = wmma_bf16(ca, frag_b_lds(Nc, 72, kbn, 48), cp3);
    }
    v8f cps[4] = {cp0, cp1, cp2, cp3};

    v8f sg[4];
    const int kb = lm + 16*half;
    #pragma unroll
    for (int g = 0; g < 4; ++g){
      v16bf bf = frag_b_lds(kTc, 72, kb, g*16);
      v8f z = {};
      v8f s = wmma_bf16(qf, bf, z);
      #pragma unroll
      for (int r = 0; r < 8; ++r) s[r] = s[r]*norm + cps[g][r];
      sg[g] = s;
    }
    #pragma unroll
    for (int r = 0; r < 8; ++r){
      float mx = mrun[r];
      #pragma unroll
      for (int g = 0; g < 4; ++g) mx = fmaxf(mx, sg[g][r]);
      #pragma unroll
      for (int off = 8; off >= 1; off >>= 1) mx = fmaxf(mx, __shfl_xor(mx, off, 32));
      float sc = __expf(mrun[r] - mx);
      float rs = 0.f;
      #pragma unroll
      for (int g = 0; g < 4; ++g){ float pw = __expf(sg[g][r] - mx); sg[g][r] = pw; rs += pw; }
      #pragma unroll
      for (int off = 8; off >= 1; off >>= 1) rs += __shfl_xor(rs, off, 32);
      lrun[r] = lrun[r]*sc + rs;
      mrun[r] = mx;
      o0[r] *= sc; o1[r] *= sc;
      int Mrow = r + 8*half;
      #pragma unroll
      for (int g = 0; g < 4; ++g) PshW[Mrow*72 + g*16 + lm] = f2bf(sg[g][r]);
    }
    __syncthreads();
    #pragma unroll
    for (int kkc = 0; kkc < 2; ++kkc){
      v16bf pa  = frag_a_lds(PshW, 72, lm, kkc*32, half);
      int   kb2 = kkc*32 + lm + 16*half;
      v16bf vb0 = frag_b_lds(vc, 40, kb2, 0);
      v16bf vb1 = frag_b_lds(vc, 40, kb2, 16);
      o0 = wmma_bf16(pa, vb0, o0);
      o1 = wmma_bf16(pa, vb1, o1);
    }
    __syncthreads();   // all reads of buf p done before it is restaged
  }
  #pragma unroll
  for (int r = 0; r < 8; ++r){
    int er = er0 + r + 8*half;
    if (er < Ee){
      float inv = 1.f / lrun[r];
      unsigned short* dst = eO + ((long)b*Ee + er)*Dd + h*HDd;
      dst[lm]      = f2bf(o0[r]*inv);
      dst[16 + lm] = f2bf(o1[r]*inv);
    }
  }
}

// ---------------------------------------------------------------------------
extern "C" void kernel_launch(void* const* d_in, const int* in_sizes, int n_in,
                              void* d_out, int out_size, void* d_ws, size_t ws_size,
                              hipStream_t stream)
{
  const float* node_feats = (const float*)d_in[0];
  const float* edge_feats = (const float*)d_in[1];
  const float* images     = (const float*)d_in[2];
  const float* rel_dist   = (const float*)d_in[3];
  const float* cbm        = (const float*)d_in[4];
  const float* W_nq   = (const float*)d_in[5];
  const float* b_nq   = (const float*)d_in[6];
  const float* W_eq   = (const float*)d_in[7];
  const float* b_eq   = (const float*)d_in[8];
  const float* W_kv   = (const float*)d_in[9];
  const float* b_kv   = (const float*)d_in[10];
  const float* W_no   = (const float*)d_in[11];
  const float* b_no   = (const float*)d_in[12];
  const float* W_eo   = (const float*)d_in[13];
  const float* b_eo   = (const float*)d_in[14];
  const float* W_cpb1 = (const float*)d_in[15];
  const float* W_cpb2 = (const float*)d_in[16];
  (void)in_sizes; (void)n_in; (void)out_size; (void)ws_size;

  // workspace carve (bf16 intermediates): ~100 MB total
  unsigned short* ws = (unsigned short*)d_ws;
  size_t off = 0;
  unsigned short* nq   = ws + off; off += (size_t)Bb*Nn*Dd;      // 307200
  unsigned short* eqp  = ws + off; off += (size_t)Bb*Ee*Dd;      // 614400
  unsigned short* kv   = ws + off; off += (size_t)Bb*Ll*2*Dd;    // 8.4M
  unsigned short* ncpb = ws + off; off += (size_t)Bb*Hh*Nn*Ll;   // 39.3M
  unsigned short* nO   = ws + off; off += (size_t)Bb*Nn*Dd;
  unsigned short* eO   = ws + off; off += (size_t)Bb*Ee*Dd;

  float* node_out = (float*)d_out;
  float* edge_out = node_out + (size_t)Bb*Nn*Dd;

  dim3 blk(256);
  size_t smemG = (size_t)(16*264 + 256*136)*2;   // 78 KB

  // Q/K/V projections: fp32 in -> bf16 out
  gemm_bias_kernel<0,1><<<dim3((Bb*Nn)/16, Dd/128),     blk, smemG, stream>>>(node_feats, W_nq, b_nq, nq,  Bb*Nn, Dd);
  gemm_bias_kernel<0,1><<<dim3((Bb*Ee)/16, Dd/128),     blk, smemG, stream>>>(edge_feats, W_eq, b_eq, eqp, Bb*Ee, Dd);
  gemm_bias_kernel<0,1><<<dim3((Bb*Ll)/16, (2*Dd)/128), blk, smemG, stream>>>(images,     W_kv, b_kv, kv,  Bb*Ll, 2*Dd);

  // fused CPB MLP -> node_cpb (B,H,N,L) bf16
  size_t smem1 = (size_t)(128*520 + 512*24)*2 + (size_t)(CPBH + CPBH + 128 + 128)*4;
  cpb_node_kernel<<<dim3(Ll/128, Nn, Bb), blk, smem1, stream>>>(rel_dist, W_cpb1, W_cpb2, ncpb);

  // flash attention (node), and edge with fused cbm@node_cpb
  attn_node_kernel<<<dim3((Nn + 127)/128, Hh, Bb), blk, 0, stream>>>(nq, kv, ncpb, nO);
  size_t smem3 = (size_t)(128*328 + 2*320*72 + 2*32*72 + 2*64*40 + 8*16*72)*2;  // ~209 KB
  attn_edge_kernel<<<dim3((Ee + 127)/128, Hh, Bb), blk, smem3, stream>>>(eqp, kv, ncpb, cbm, eO);

  // output projections: bf16 in -> fp32 out (d_out)
  gemm_bias_kernel<1,0><<<dim3((Bb*Nn)/16, Dd/128), blk, smemG, stream>>>(nO, W_no, b_no, node_out, Bb*Nn, Dd);
  gemm_bias_kernel<1,0><<<dim3((Bb*Ee)/16, Dd/128), blk, smemG, stream>>>(eO, W_eo, b_eo, edge_out, Bb*Ee, Dd);
}
